// MultiHeadLatentAttention_68564857914083
// MI455X (gfx1250) — matrix operand order repaired and checked by reading, not compiled, not run on hardware
//
#include <hip/hip_runtime.h>
#include <hip/hip_bf16.h>

// MI455X (gfx1250) Multi-Head Latent Attention, mixed precision:
// f32 in/out, f16 WMMA compute with f32 accumulation, flash-attention
// inner loop (no [B,H,S,S] materialization). Big GEMMs stage weight tiles
// into LDS with async global->LDS copies (ASYNCcnt), double-buffered.

typedef __attribute__((ext_vector_type(16))) _Float16 v16h;
typedef __attribute__((ext_vector_type(8)))  _Float16 v8h;
typedef __attribute__((ext_vector_type(8)))  float    v8f;

#define B_     4
#define S_     2048
#define D_     1024
#define H_     16
#define DEPTH_ 64
#define L_     32

// ---------------- WMMA fragment loaders (wave32, contiguous k order for both A and B) -------

// A (16x32 f16, row-major [M,K]): lane L holds row m=L&15;
// half i -> k = (L>>4)*16 + i  (16 contiguous halves)
static __device__ __forceinline__ v16h load_a_frag(const _Float16* __restrict__ A,
                                                   int lda, int row0, int kk) {
    int lane = threadIdx.x & 31;
    int m    = lane & 15;
    int kb   = (lane >> 4) * 16;
    const _Float16* p = A + (size_t)(row0 + m) * lda + kk + kb;
    v8h lo = *(const v8h*)p;          // k = kk+kb .. kk+kb+7
    v8h hi = *(const v8h*)(p + 8);    // k = kk+kb+8 .. +7
    v16h r;
#pragma unroll
    for (int i = 0; i < 8; ++i) { r[i] = lo[i]; r[i + 8] = hi[i]; }
    return r;
}

// B (32x16 f16) loaded from a TRANSPOSED [N,K] row-major buffer:
// lane L holds col n=L&15; half i -> k = (L>>4)*16 + i  (16 contiguous halves)
static __device__ __forceinline__ v16h load_bt_frag(const _Float16* __restrict__ Bt,
                                                    int ldk, int col0, int kk) {
    int lane = threadIdx.x & 31;
    int n    = lane & 15;
    int kb   = (lane >> 4) * 16;
    return *(const v16h*)(Bt + (size_t)(col0 + n) * ldk + kk + kb);
}

// ---------------- elementwise converts -------------------------------------

__global__ void cvt_f32_f16(const float* __restrict__ in, _Float16* __restrict__ out, int n) {
    int i = blockIdx.x * blockDim.x + threadIdx.x;
    if (i < n) out[i] = (_Float16)in[i];
}

// out[n*K + k] = in[k*N + n]   (weight transpose + f16 convert)
__global__ void transpose_cvt(const float* __restrict__ in, _Float16* __restrict__ out,
                              int K, int N) {
    int i = blockIdx.x * blockDim.x + threadIdx.x;
    if (i < K * N) {
        int k = i % K, n = i / K;
        out[i] = (_Float16)in[(size_t)k * N + n];
    }
}

// ---------------- store helper for GEMM outputs ----------------------------
// MODE 0: f16 row-major [M,N]
// MODE 1: f16 head-split   q/k : [B,H,S,DEPTH]
// MODE 2: f16 head-split-T v   : [B,H,DEPTH,S]
// MODE 3: f32 row-major (final output)
template <int MODE>
static __device__ __forceinline__ void store_c(void* Cout, int m, int n, int N, float val) {
    if (MODE == 0) {
        ((_Float16*)Cout)[(size_t)m * N + n] = (_Float16)val;
    } else if (MODE == 1) {
        int b = m / S_, s = m % S_, h = n >> 6, d = n & 63;
        ((_Float16*)Cout)[(((size_t)b * H_ + h) * S_ + s) * DEPTH_ + d] = (_Float16)val;
    } else if (MODE == 2) {
        int b = m / S_, s = m % S_, h = n >> 6, d = n & 63;
        ((_Float16*)Cout)[(((size_t)b * H_ + h) * DEPTH_ + d) * S_ + s] = (_Float16)val;
    } else {
        ((float*)Cout)[(size_t)m * N + n] = val;
    }
}

// ---------------- simple direct-load WMMA GEMM (small-K latent stage) ------
template <int NT, int MODE>
__global__ void wmma_gemm(const _Float16* __restrict__ A,
                          const _Float16* __restrict__ Bt,
                          const float* __restrict__ bias,
                          void* __restrict__ Cout,
                          int M, int N, int K) {
    int wave = blockIdx.x * (blockDim.x >> 5) + (threadIdx.x >> 5);
    int lane = threadIdx.x & 31;
    int ntw  = N / (16 * NT);
    int mt   = wave / ntw;
    int nt0  = (wave % ntw) * NT;
    if (mt * 16 >= M) return;
    int row0 = mt * 16;

    v8f acc[NT] = {};
    for (int kk = 0; kk < K; kk += 32) {
        v16h a = load_a_frag(A, K, row0, kk);
#pragma unroll
        for (int t = 0; t < NT; ++t) {
            v16h b = load_bt_frag(Bt, K, (nt0 + t) * 16, kk);
            acc[t] = __builtin_amdgcn_wmma_f32_16x16x32_f16(
                false, a, false, b, (short)0, acc[t], false, false);
        }
    }

    int nlo = lane & 15;
    int mhi = (lane >> 4) * 8;
#pragma unroll
    for (int t = 0; t < NT; ++t) {
        int n = (nt0 + t) * 16 + nlo;
        float bval = bias ? bias[n] : 0.f;
#pragma unroll
        for (int r = 0; r < 8; ++r)
            store_c<MODE>(Cout, row0 + mhi + r, n, N, acc[t][r] + bval);
    }
}

// ---------------- LDS-staged WMMA GEMM (big projections) -------------------
// Workgroup = 8 waves = 128 rows x 64 cols. The 64x32 f16 weight chunk (4 KB)
// is staged into LDS with one global_load_async_to_lds_b128 per wave
// (8 waves x 32 lanes x 16B = 4 KB), double-buffered; all 8 waves reuse it.
template <int MODE>
__global__ void wmma_gemm_lds(const _Float16* __restrict__ A,
                              const _Float16* __restrict__ Bt,
                              const float* __restrict__ bias,
                              void* __restrict__ Cout,
                              int M, int N, int K) {
    __shared__ __align__(64) _Float16 btile[2][64 * 32];   // [n][k], 64B rows
    int wlocal = threadIdx.x >> 5;
    int lane   = threadIdx.x & 31;
    int ngrp   = N / 64;
    int mb     = blockIdx.x / ngrp;
    int nt0    = (blockIdx.x % ngrp) * 4;       // four 16-col tiles
    int row0   = mb * 128 + wlocal * 16;

    // This thread's 16-byte unit of the staged B chunk.
    int u  = wlocal * 32 + lane;                // 0..255
    int bn = u >> 2;                            // B row within N-group (0..63)
    int bk = (u & 3) * 8;                       // half offset within row
    const _Float16* gp = Bt + (size_t)(nt0 * 16 + bn) * K + bk;
    unsigned lds_u = (unsigned)(uintptr_t)(&btile[0][0]) + (unsigned)u * 16;
    const unsigned lds_stride = 64 * 32 * 2;    // bytes per buffer

    // prologue: stage chunk 0 into buffer 0 (ASYNCcnt-tracked)
    asm volatile("global_load_async_to_lds_b128 %0, %1, off"
                 :: "v"(lds_u), "v"(gp) : "memory");

    v8f acc[4] = {};
    const int nchunks = K / 32;
    for (int ci = 0; ci < nchunks; ++ci) {
        int kk = ci * 32;
        if (ci + 1 < nchunks) {
            unsigned dst = lds_u + (unsigned)((ci + 1) & 1) * lds_stride;
            const _Float16* g = gp + (kk + 32);
            asm volatile("global_load_async_to_lds_b128 %0, %1, off"
                         :: "v"(dst), "v"(g) : "memory");
            asm volatile("s_wait_asynccnt 1" ::: "memory");   // chunk ci done
        } else {
            asm volatile("s_wait_asynccnt 0" ::: "memory");
        }
        __syncthreads();                        // everyone's portion landed

        const _Float16* bt = &btile[ci & 1][0];
        v16h a = load_a_frag(A, K, row0, kk);
        int kb2 = (lane >> 4) * 16;
        int nlo = lane & 15;
#pragma unroll
        for (int t = 0; t < 4; ++t) {
            v16h b = *(const v16h*)(bt + (t * 16 + nlo) * 32 + kb2);
            acc[t] = __builtin_amdgcn_wmma_f32_16x16x32_f16(
                false, a, false, b, (short)0, acc[t], false, false);
        }
        __syncthreads();                        // safe to overwrite this buffer
    }

    int nlo = lane & 15;
    int mhi = (lane >> 4) * 8;
#pragma unroll
    for (int t = 0; t < 4; ++t) {
        int n = (nt0 + t) * 16 + nlo;
        float bval = bias ? bias[n] : 0.f;
#pragma unroll
        for (int r = 0; r < 8; ++r)
            store_c<MODE>(Cout, row0 + mhi + r, n, N, acc[t][r] + bval);
    }
}

// ---------------- flash attention over latent scores -----------------------
// lq,lk: [B*H, S, L] f16 ; vt: [B*H, DEPTH, S] f16 ; ctx: [B, S, H, DEPTH] f16
// One wave owns a 16-query tile; online softmax over 32 keys per iteration.
__global__ void mla_attention(const _Float16* __restrict__ lq,
                              const _Float16* __restrict__ lk,
                              const _Float16* __restrict__ vt,
                              _Float16* __restrict__ ctx) {
    __shared__ __align__(64) _Float16 plds_all[8][16 * 32]; // per-wave P staging
    int wlocal = threadIdx.x >> 5;
    int wave   = blockIdx.x * (blockDim.x >> 5) + wlocal;
    int lane   = threadIdx.x & 31;
    int bh     = wave >> 7;          // S_/16 = 128 query tiles per (b,h)
    int qt     = wave & 127;
    const _Float16* lqp = lq + (size_t)bh * S_ * L_;
    const _Float16* lkp = lk + (size_t)bh * S_ * L_;
    const _Float16* vp  = vt + (size_t)bh * DEPTH_ * S_;
    _Float16* plds = plds_all[wlocal];

    const float scale = 0.17677669529663687f;   // 1/sqrt(L)
    v16h aq = load_a_frag(lqp, L_, qt * 16, 0); // K = L = 32: single A fragment

    float m_run[8], l_run[8];
    v8f acc[4] = {};
#pragma unroll
    for (int r = 0; r < 8; ++r) { m_run[r] = -3.0e30f; l_run[r] = 0.f; }

    int mrow = (lane >> 4) * 8;   // C-layout row base for this lane half
    int ncol = lane & 15;         // C-layout column

    for (int kb = 0; kb < S_; kb += 32) {
        v16h b0 = load_bt_frag(lkp, L_, kb, 0);       // keys kb..kb+15
        v16h b1 = load_bt_frag(lkp, L_, kb + 16, 0);  // keys kb+16..kb+31
        v8f z = {};
        v8f s0 = __builtin_amdgcn_wmma_f32_16x16x32_f16(false, aq, false, b0, (short)0, z, false, false);
        v8f s1 = __builtin_amdgcn_wmma_f32_16x16x32_f16(false, aq, false, b1, (short)0, z, false, false);

        float rmax[8];
#pragma unroll
        for (int r = 0; r < 8; ++r) {
            s0[r] *= scale; s1[r] *= scale;
            rmax[r] = fmaxf(s0[r], s1[r]);
        }
#pragma unroll
        for (int mask = 1; mask <= 8; mask <<= 1)
#pragma unroll
            for (int r = 0; r < 8; ++r)
                rmax[r] = fmaxf(rmax[r], __shfl_xor(rmax[r], mask, 32));

        float corr[8], rsum[8];
#pragma unroll
        for (int r = 0; r < 8; ++r) {
            float mn = fmaxf(m_run[r], rmax[r]);
            corr[r]  = __expf(m_run[r] - mn);
            m_run[r] = mn;
            float p0 = __expf(s0[r] - mn);
            float p1 = __expf(s1[r] - mn);
            rsum[r]  = p0 + p1;
            plds[(mrow + r) * 32 + ncol]      = (_Float16)p0;   // C-layout -> [m][k]
            plds[(mrow + r) * 32 + 16 + ncol] = (_Float16)p1;
        }
#pragma unroll
        for (int mask = 1; mask <= 8; mask <<= 1)
#pragma unroll
            for (int r = 0; r < 8; ++r)
                rsum[r] += __shfl_xor(rsum[r], mask, 32);
#pragma unroll
        for (int r = 0; r < 8; ++r) l_run[r] = l_run[r] * corr[r] + rsum[r];
#pragma unroll
        for (int c = 0; c < 4; ++c)
#pragma unroll
            for (int r = 0; r < 8; ++r) acc[c][r] *= corr[r];

        // reload P as an A fragment (wave-private LDS; DS ops are in-order)
        v16h pa;
        {
            int m   = lane & 15;
            int kb2 = (lane >> 4) * 16;
            const _Float16* pp = plds + m * 32 + kb2;
            v8h lo = *(const v8h*)pp;
            v8h hi = *(const v8h*)(pp + 8);
#pragma unroll
            for (int i = 0; i < 8; ++i) { pa[i] = lo[i]; pa[i + 8] = hi[i]; }
        }
#pragma unroll
        for (int c = 0; c < 4; ++c) {
            v16h bvf = load_bt_frag(vp, S_, c * 16, kb);  // Vt[d, key]: contiguous keys
            acc[c] = __builtin_amdgcn_wmma_f32_16x16x32_f16(false, pa, false, bvf, (short)0, acc[c], false, false);
        }
    }

    int b = bh >> 4, h = bh & 15;
#pragma unroll
    for (int r = 0; r < 8; ++r) {
        float inv = 1.f / l_run[r];
        int s = qt * 16 + mrow + r;
#pragma unroll
        for (int c = 0; c < 4; ++c) {
            int d = c * 16 + ncol;
            ctx[(((size_t)b * S_ + s) * H_ + h) * DEPTH_ + d] = (_Float16)(acc[c][r] * inv);
        }
    }
}

// ---------------- host orchestration ---------------------------------------

extern "C" void kernel_launch(void* const* d_in, const int* in_sizes, int n_in,
                              void* d_out, int out_size, void* d_ws, size_t ws_size,
                              hipStream_t stream) {
    const float* queries = (const float*)d_in[0];
    const float* keys    = (const float*)d_in[1];
    const float* values  = (const float*)d_in[2];
    const float* Wq   = (const float*)d_in[3];
    const float* b_q  = (const float*)d_in[4];
    const float* Wk   = (const float*)d_in[5];
    const float* b_k  = (const float*)d_in[6];
    const float* Wv   = (const float*)d_in[7];
    const float* b_v  = (const float*)d_in[8];
    const float* Wlq  = (const float*)d_in[9];
    const float* b_lq = (const float*)d_in[10];
    const float* Wlk  = (const float*)d_in[11];
    const float* b_lk = (const float*)d_in[12];
    const float* Wo   = (const float*)d_in[13];
    const float* b_o  = (const float*)d_in[14];

    const int TOK = B_ * S_;            // 8192
    const int NX  = TOK * D_;           // 8,388,608 elems
    char* ws = (char*)d_ws;
    size_t off = 0;
    auto alloc = [&](size_t bytes) -> void* {
        void* p = ws + off;
        off += (bytes + 255) & ~(size_t)255;
        return p;
    };
    _Float16* xq    = (_Float16*)alloc((size_t)NX * 2);
    _Float16* xk    = (_Float16*)alloc((size_t)NX * 2);
    _Float16* xv    = (_Float16*)alloc((size_t)NX * 2);
    _Float16* wq_t  = (_Float16*)alloc((size_t)D_ * D_ * 2);
    _Float16* wk_t  = (_Float16*)alloc((size_t)D_ * D_ * 2);
    _Float16* wv_t  = (_Float16*)alloc((size_t)D_ * D_ * 2);
    _Float16* wo_t  = (_Float16*)alloc((size_t)D_ * D_ * 2);
    _Float16* wlq_t = (_Float16*)alloc((size_t)DEPTH_ * L_ * 2);
    _Float16* wlk_t = (_Float16*)alloc((size_t)DEPTH_ * L_ * 2);
    _Float16* qh    = (_Float16*)alloc((size_t)NX * 2);   // [B,H,S,DEPTH]
    _Float16* kh    = (_Float16*)alloc((size_t)NX * 2);   // [B,H,S,DEPTH]
    _Float16* vt    = (_Float16*)alloc((size_t)NX * 2);   // [B,H,DEPTH,S]
    _Float16* lqb   = (_Float16*)alloc((size_t)B_ * H_ * S_ * L_ * 2);
    _Float16* lkb   = (_Float16*)alloc((size_t)B_ * H_ * S_ * L_ * 2);
    _Float16* ctx   = (_Float16*)alloc((size_t)NX * 2);   // [B,S,H,DEPTH]

    dim3 tpb(256);

    // 1) input converts
    cvt_f32_f16<<<NX / 256, tpb, 0, stream>>>(queries, xq, NX);
    cvt_f32_f16<<<NX / 256, tpb, 0, stream>>>(keys,    xk, NX);
    cvt_f32_f16<<<NX / 256, tpb, 0, stream>>>(values,  xv, NX);

    // 2) weight transpose + convert
    transpose_cvt<<<(D_ * D_) / 256, tpb, 0, stream>>>(Wq, wq_t, D_, D_);
    transpose_cvt<<<(D_ * D_) / 256, tpb, 0, stream>>>(Wk, wk_t, D_, D_);
    transpose_cvt<<<(D_ * D_) / 256, tpb, 0, stream>>>(Wv, wv_t, D_, D_);
    transpose_cvt<<<(D_ * D_) / 256, tpb, 0, stream>>>(Wo, wo_t, D_, D_);
    transpose_cvt<<<(DEPTH_ * L_ + 255) / 256, tpb, 0, stream>>>(Wlq, wlq_t, DEPTH_, L_);
    transpose_cvt<<<(DEPTH_ * L_ + 255) / 256, tpb, 0, stream>>>(Wlk, wlk_t, DEPTH_, L_);

    // 3) Q/K/V projections: [8192,1024] @ [1024,1024]
    //    128x64 per 8-wave block, async-LDS staged weights
    {
        int grids = (TOK / 128) * (D_ / 64);   // 64 * 16 = 1024 blocks
        wmma_gemm_lds<1><<<grids, tpb, 0, stream>>>(xq, wq_t, b_q, qh, TOK, D_, D_);
        wmma_gemm_lds<1><<<grids, tpb, 0, stream>>>(xk, wk_t, b_k, kh, TOK, D_, D_);
        wmma_gemm_lds<2><<<grids, tpb, 0, stream>>>(xv, wv_t, b_v, vt, TOK, D_, D_);
    }

    // 4) latent projections: [B*H*S, 64] @ [64, 32]; 16x32 tile per wave
    {
        int M = B_ * H_ * S_;                 // 131072
        int waves = (M / 16) * (L_ / 32);     // 8192
        int grids = waves / 8;
        wmma_gemm<2, 0><<<grids, tpb, 0, stream>>>(qh, wlq_t, b_lq, lqb, M, L_, DEPTH_);
        wmma_gemm<2, 0><<<grids, tpb, 0, stream>>>(kh, wlk_t, b_lk, lkb, M, L_, DEPTH_);
    }

    // 5) flash attention: one wave per 16-query tile per (b,h)
    {
        int waves = B_ * H_ * (S_ / 16);      // 8192
        mla_attention<<<waves / 8, tpb, 0, stream>>>(lqb, lkb, vt, ctx);
    }

    // 6) output projection -> f32 d_out (async-LDS staged weights)
    {
        int grids = (TOK / 128) * (D_ / 64);
        wmma_gemm_lds<3><<<grids, tpb, 0, stream>>>(ctx, wo_t, b_o, d_out, TOK, D_, D_);
    }
}
